// MultiScaleElmanCell_6399501271091
// MI455X (gfx1250) — compile-verified, matrix operand order checked by reading, not run on hardware
//
#include <hip/hip_runtime.h>
#include <math.h>

#define T_STEPS 512
#define BATCH   16
#define DIM     1024
#define NBANK   4
#define NWG     16
#define COLS    64            // DIM / NWG columns per workgroup
#define WSTR    1034          // padded LDS row stride (f16 elems) -> conflict-free WMMA B reads
#define ZROW    (DIM * (1 + NBANK))   // 5120

typedef __attribute__((ext_vector_type(16))) _Float16 v16h;
typedef __attribute__((ext_vector_type(8)))  float    v8f;

union AF { unsigned u[8]; _Float16 e[16]; v16h v; };

__device__ __forceinline__ float silu_f(float x) {
  return x / (1.0f + __expf(-x));
}

__device__ __forceinline__ float block_reduce(float val, float* red) {
  const int tid = threadIdx.x;
  red[tid] = val;
  __syncthreads();
  for (int s = 128; s > 0; s >>= 1) {
    if (tid < s) red[tid] += red[tid + s];
    __syncthreads();
  }
  float r = red[0];
  __syncthreads();
  return r;
}

// ---------------------------------------------------------------------------
// Kernel 1: spectral normalization scale for W_h (3-step power iteration).
// Writes s = 0.99 / (sigma + 1e-8) to wsf[0].
// ---------------------------------------------------------------------------
__global__ __launch_bounds__(256) void msec_spectral(
    const float* __restrict__ Wh, const float* __restrict__ u0,
    float* __restrict__ wsf) {
  __shared__ float u[DIM];
  __shared__ float v[DIM];
  __shared__ float red[256];
  const int tid = threadIdx.x;

  // u = u0 / ||u0||
  float uv[4];
  float p = 0.0f;
#pragma unroll
  for (int q = 0; q < 4; ++q) {
    uv[q] = u0[tid + q * 256];
    p += uv[q] * uv[q];
  }
  float nrm = sqrtf(block_reduce(p, red));
#pragma unroll
  for (int q = 0; q < 4; ++q) u[tid + q * 256] = uv[q] / nrm;
  __syncthreads();

  float sumsq_u = 0.0f;
  for (int it = 0; it < 3; ++it) {
    // v = Wh^T @ u  (coalesced: at step i, threads read consecutive columns)
    float vloc[4];
    float pv = 0.0f;
#pragma unroll
    for (int q = 0; q < 4; ++q) {
      const int j = tid + q * 256;
      float s = 0.0f;
      for (int i = 0; i < DIM; ++i) s += Wh[(size_t)i * DIM + j] * u[i];
      vloc[q] = s;
      pv += s * s;
    }
    float nv = sqrtf(block_reduce(pv, red)) + 1e-8f;
#pragma unroll
    for (int q = 0; q < 4; ++q) v[tid + q * 256] = vloc[q] / nv;
    __syncthreads();

    // u = Wh @ v
    float uloc[4];
    float pu = 0.0f;
#pragma unroll
    for (int q = 0; q < 4; ++q) {
      const int i = tid + q * 256;
      float s = 0.0f;
      for (int j = 0; j < DIM; ++j) s += Wh[(size_t)i * DIM + j] * v[j];
      uloc[q] = s;
      pu += s * s;
    }
    sumsq_u = block_reduce(pu, red);
    float nu = sqrtf(sumsq_u) + 1e-8f;
#pragma unroll
    for (int q = 0; q < 4; ++q) u[tid + q * 256] = uloc[q] / nu;
    __syncthreads();
  }

  if (tid == 0) {
    // u_final = u_pre/(||u_pre||+eps) with u_pre = Wh@v  =>
    // sigma = |u_final . u_pre| = sumsq / (sqrt(sumsq)+eps)
    float nraw = sqrtf(sumsq_u);
    float sigma = sumsq_u / (nraw + 1e-8f);
    wsf[0] = 0.99f / (sigma + 1e-8f);
  }
}

// ---------------------------------------------------------------------------
// Software grid barrier over NWG persistent workgroups.
// ---------------------------------------------------------------------------
__device__ __forceinline__ void grid_sync(unsigned* cnt, unsigned step) {
  __syncthreads();
  if (threadIdx.x == 0) {
    __threadfence();
    __hip_atomic_fetch_add(cnt, 1u, __ATOMIC_RELEASE, __HIP_MEMORY_SCOPE_AGENT);
    const unsigned target = (unsigned)NWG * step;
    while (__hip_atomic_load(cnt, __ATOMIC_ACQUIRE, __HIP_MEMORY_SCOPE_AGENT) < target)
      __builtin_amdgcn_s_sleep(1);
    __threadfence();
  }
  __syncthreads();
}

// WMMA 16x16x32 f16: K offset inside a 32-chunk for VGPR p, lane-group g.
__device__ __forceinline__ int koff(int p, int g) {
  return ((p >> 2) << 4) + (g << 3) + ((p & 3) << 1);
}

// ---------------------------------------------------------------------------
// Kernel 2: persistent recurrence. 16 WGs x 256 threads; each WG owns 64
// output columns and keeps its f16 weight slices resident in LDS.
// ---------------------------------------------------------------------------
__global__ __launch_bounds__(256, 1) void msec_recurrent(
    const float* __restrict__ x,  const float* __restrict__ z,
    const float* __restrict__ h0, const float* __restrict__ m0,
    const float* __restrict__ Wx, const float* __restrict__ Wh,
    const float* __restrict__ bvec, const float* __restrict__ a,
    float* __restrict__ out, char* __restrict__ ws) {
  extern __shared__ char smem[];
  _Float16* WhL  = (_Float16*)smem;                              // [COLS][WSTR]
  _Float16* WxL  = (_Float16*)(smem + (size_t)COLS * WSTR * 2);  // [COLS][WSTR]
  float*    accl = (float*)(smem + 2u * (size_t)COLS * WSTR * 2);// [4][16][16]
  float*    biasl = accl + 4 * 256;                              // [COLS]

  const int wg  = blockIdx.x;
  const int tid = threadIdx.x;

  const float scale = *(const float*)ws;          // from msec_spectral
  unsigned* cnt = (unsigned*)(ws + 16);
  _Float16* hbuf0 = (_Float16*)(ws + 256);                       // [16][1024] f16
  _Float16* hbuf1 = (_Float16*)(ws + 256 + BATCH * DIM * 2);

  // ---- Stage weight slices into LDS (f32 -> f16, Wh scaled) ----
  for (int idx = tid; idx < COLS * 256; idx += 256) {
    const int c  = idx >> 8;
    const int k4 = (idx & 255) << 2;
    const int j  = wg * COLS + c;
    float4 wh = *(const float4*)(Wh + (size_t)j * DIM + k4);
    float4 wx = *(const float4*)(Wx + (size_t)j * DIM + k4);
    _Float16* dh = WhL + (size_t)c * WSTR + k4;
    _Float16* dx = WxL + (size_t)c * WSTR + k4;
    dh[0] = (_Float16)(wh.x * scale); dh[1] = (_Float16)(wh.y * scale);
    dh[2] = (_Float16)(wh.z * scale); dh[3] = (_Float16)(wh.w * scale);
    dx[0] = (_Float16)wx.x; dx[1] = (_Float16)wx.y;
    dx[2] = (_Float16)wx.z; dx[3] = (_Float16)wx.w;
  }
  if (tid < COLS) biasl[tid] = bvec[wg * COLS + tid];

  // ---- Per-thread element ownership: batch row bi, 4 consecutive cols ----
  const int bi = tid >> 4;
  const int c0 = (tid & 15) << 2;
  const int jg = wg * COLS + c0;

  // Stage h0 slice into ping-pong buffer 0.
  {
    float4 hv = *(const float4*)(h0 + (size_t)bi * DIM + jg);
    _Float16* d = hbuf0 + (size_t)bi * DIM + jg;
    d[0] = (_Float16)hv.x; d[1] = (_Float16)hv.y;
    d[2] = (_Float16)hv.z; d[3] = (_Float16)hv.w;
  }

  // alpha = sigmoid(a), m state in registers.
  float alr[NBANK][4], mreg[NBANK][4];
#pragma unroll
  for (int nb = 0; nb < NBANK; ++nb) {
    float4 av = *(const float4*)(a + (size_t)nb * DIM + jg);
    alr[nb][0] = 1.0f / (1.0f + __expf(-av.x));
    alr[nb][1] = 1.0f / (1.0f + __expf(-av.y));
    alr[nb][2] = 1.0f / (1.0f + __expf(-av.z));
    alr[nb][3] = 1.0f / (1.0f + __expf(-av.w));
    float4 mv = *(const float4*)(m0 + ((size_t)nb * BATCH + bi) * DIM + jg);
    mreg[nb][0] = mv.x; mreg[nb][1] = mv.y; mreg[nb][2] = mv.z; mreg[nb][3] = mv.w;
  }

  grid_sync(cnt, 1);   // all slices of h0 staged, weights resident

  const int wave = tid >> 5;
  const int lane = tid & 31;
  const int g    = (lane >> 4) & 1;
  const int mn   = lane & 15;          // = A-matrix row (batch) / B,D column
  const v8f zero8 = {0.f, 0.f, 0.f, 0.f, 0.f, 0.f, 0.f, 0.f};
  float hreg[4] = {0.f, 0.f, 0.f, 0.f};

  for (int t = 0; t < T_STEPS; ++t) {
    const _Float16* hcur = (t & 1) ? hbuf1 : hbuf0;
    _Float16*       hnxt = (t & 1) ? hbuf0 : hbuf1;

    // zero the cross-wave accumulator tiles
    *(float4*)(accl + (tid << 2)) = make_float4(0.f, 0.f, 0.f, 0.f);
    __syncthreads();

    // ---- GEMM: each wave covers K range [wave*128, wave*128+128) ----
    v8f acc[4] = {zero8, zero8, zero8, zero8};
    const float*    xt = x + (size_t)t * (BATCH * DIM) + (size_t)mn * DIM;
    const _Float16* hr = hcur + (size_t)mn * DIM;

#pragma unroll
    for (int q = 0; q < 4; ++q) {
      const int k0 = wave * 128 + q * 32;
      AF ah, ax;
#pragma unroll
      for (int p = 0; p < 8; ++p) {
        const int k = k0 + koff(p, g);
        ah.u[p] = *(const unsigned*)(hr + k);          // f16 pair
        float2 fx = *(const float2*)(xt + k);          // f32 pair -> f16
        ax.e[2 * p]     = (_Float16)fx.x;
        ax.e[2 * p + 1] = (_Float16)fx.y;
      }
#pragma unroll
      for (int nt = 0; nt < 4; ++nt) {
        const int c = nt * 16 + mn;
        AF bh, bx;
#pragma unroll
        for (int p = 0; p < 8; ++p) {
          const int k = k0 + koff(p, g);
          bh.u[p] = *(const unsigned*)(WhL + (size_t)c * WSTR + k);
          bx.u[p] = *(const unsigned*)(WxL + (size_t)c * WSTR + k);
        }
        acc[nt] = __builtin_amdgcn_wmma_f32_16x16x32_f16(
            false, ah.v, false, bh.v, (short)0, acc[nt], false, false);
        acc[nt] = __builtin_amdgcn_wmma_f32_16x16x32_f16(
            false, ax.v, false, bx.v, (short)0, acc[nt], false, false);
      }
    }

    // ---- cross-wave K reduction into LDS ----
#pragma unroll
    for (int nt = 0; nt < 4; ++nt) {
#pragma unroll
      for (int r = 0; r < 8; ++r) {
        const int mrow = g * 8 + r;
        atomicAdd(accl + nt * 256 + mrow * 16 + mn, acc[nt][r]);
      }
    }
    __syncthreads();

    // ---- pointwise: tanh, EMA banks, SiLU-gated output ----
    float zg[5][4];
    const float* zt = z + ((size_t)t * BATCH + bi) * ZROW + jg;
#pragma unroll
    for (int gg = 0; gg < 5; ++gg) {
      float4 zv = *(const float4*)(zt + gg * DIM);
      zg[gg][0] = zv.x; zg[gg][1] = zv.y; zg[gg][2] = zv.z; zg[gg][3] = zv.w;
    }
    float o[4];
#pragma unroll
    for (int ci = 0; ci < 4; ++ci) {
      const int c = c0 + ci;
      float vacc = accl[((c >> 4) << 8) + (bi << 4) + (c & 15)] + biasl[c];
      float hn = tanhf(vacc);
      hreg[ci] = hn;
      hnxt[(size_t)bi * DIM + wg * COLS + c] = (_Float16)hn;
      float oa = hn * silu_f(zg[0][ci]);
#pragma unroll
      for (int nb = 0; nb < NBANK; ++nb) {
        const float al = alr[nb][ci];
        mreg[nb][ci] = al * mreg[nb][ci] + (1.0f - al) * hn;
        oa += mreg[nb][ci] * silu_f(zg[nb + 1][ci]);
      }
      o[ci] = oa;
    }
    *(float4*)(out + ((size_t)t * BATCH + bi) * DIM + jg) =
        make_float4(o[0], o[1], o[2], o[3]);

    // ---- prefetch next step's x (full) and our z slice while barrier drains
    if (t + 1 < T_STEPS) {
      __builtin_prefetch((const char*)(x + (size_t)(t + 1) * BATCH * DIM) + tid * 256, 0, 0);
      if (tid < 80) {
        const int gate = tid >> 4, bb = tid & 15;
        __builtin_prefetch(
            z + ((size_t)(t + 1) * BATCH + bb) * ZROW + gate * DIM + wg * COLS, 0, 0);
      }
    }

    grid_sync(cnt, (unsigned)(t + 2));
  }

  // ---- final h and m states ----
  const size_t OUT_H = (size_t)T_STEPS * BATCH * DIM;
  float* hf = out + OUT_H;
  *(float4*)(hf + (size_t)bi * DIM + jg) =
      make_float4(hreg[0], hreg[1], hreg[2], hreg[3]);
  float* mf = out + OUT_H + (size_t)BATCH * DIM;
#pragma unroll
  for (int nb = 0; nb < NBANK; ++nb) {
    *(float4*)(mf + ((size_t)nb * BATCH + bi) * DIM + jg) =
        make_float4(mreg[nb][0], mreg[nb][1], mreg[nb][2], mreg[nb][3]);
  }
}

// ---------------------------------------------------------------------------
extern "C" void kernel_launch(void* const* d_in, const int* in_sizes, int n_in,
                              void* d_out, int out_size, void* d_ws, size_t ws_size,
                              hipStream_t stream) {
  const float* x  = (const float*)d_in[0];
  const float* z  = (const float*)d_in[1];
  const float* h0 = (const float*)d_in[2];
  const float* m0 = (const float*)d_in[3];
  const float* Wx = (const float*)d_in[4];
  const float* Wh = (const float*)d_in[5];
  const float* bv = (const float*)d_in[6];
  const float* a  = (const float*)d_in[7];
  const float* u0 = (const float*)d_in[8];
  float* out = (float*)d_out;
  char*  ws  = (char*)d_ws;

  (void)in_sizes; (void)n_in; (void)out_size; (void)ws_size;

  // zero the scale slot + grid-barrier counter (graph-capturable)
  hipMemsetAsync(ws, 0, 256, stream);

  msec_spectral<<<1, 256, 0, stream>>>(Wh, u0, (float*)ws);

  const size_t SMEM = 2u * (size_t)COLS * WSTR * 2   // weight slices (f16)
                    + 4 * 256 * sizeof(float)        // accumulator tiles
                    + COLS * sizeof(float);          // bias slice
  msec_recurrent<<<NWG, 256, SMEM, stream>>>(x, z, h0, m0, Wx, Wh, bv, a, out, ws);
}
